// MultiHeadAttention_71854802862114
// MI455X (gfx1250) — matrix operand order repaired and checked by reading, not compiled, not run on hardware
//
#include <hip/hip_runtime.h>

#define B_ 8
#define C_ 1024
#define T_ 1024
#define H_ 16
#define D_ 64

typedef __attribute__((ext_vector_type(16))) __bf16 v16bf;
typedef __attribute__((ext_vector_type(8)))  float  v8f;

union BFrag { v16bf v; uint4 q[2]; };   // 32B: one WMMA bf16 A or B operand
union FAcc  { v8f   v; float f[8]; };   // 32B: one WMMA f32 C/D accumulator

__device__ __forceinline__ unsigned short f2bf(float x) {
  __bf16 h = (__bf16)x;                                   // v_cvt_pk_bf16_f32 (RNE)
  return __builtin_bit_cast(unsigned short, h);
}
__device__ __forceinline__ unsigned int f2bf2(float lo, float hi) {
  union { __bf16 h[2]; unsigned int u; } p;
  p.h[0] = (__bf16)lo; p.h[1] = (__bf16)hi;
  return p.u;
}
__device__ __forceinline__ float log1p_abs(float d) {     // arg >= 1: raw v_log_f32 ok
#if __has_builtin(__builtin_amdgcn_logf)
  return __builtin_amdgcn_logf(1.0f + __builtin_fabsf(d)) * 0.69314718056f;
#else
  return __logf(1.0f + __builtin_fabsf(d));
#endif
}

// CDNA5 async memory->LDS (ASYNCcnt-tracked), GV mode: 64-bit VGPR address.
// LDS destination = low 32 bits of the generic shared pointer (flat aperture).
__device__ __forceinline__ void async_b128(const unsigned short* g, unsigned short* l) {
  asm volatile("global_load_async_to_lds_b128 %0, %1, off"
               :: "v"((unsigned int)(unsigned long long)l),
                  "v"((unsigned long long)g)
               : "memory");
}
__device__ __forceinline__ void wait_async() {
  asm volatile("s_wait_asynccnt 0x0" ::: "memory");
}

// ---------------------------------------------------------------------------
// Kernel 0: [B][C][T] f32  ->  [B][T][C] bf16  (coalesced both sides via LDS)
// ---------------------------------------------------------------------------
__global__ __launch_bounds__(256) void transpose_bf16_kernel(
    const float* __restrict__ in, unsigned short* __restrict__ out) {
  __shared__ float tile[32][33];
  const int b = blockIdx.z;
  const int t0 = blockIdx.x * 32, c0 = blockIdx.y * 32;
  const int x = threadIdx.x, y = threadIdx.y;             // 32 x 8
  const float* src = in + ((size_t)b * C_ + c0) * T_ + t0;
#pragma unroll
  for (int i = 0; i < 32; i += 8)
    tile[y + i][x] = src[(size_t)(y + i) * T_ + x];
  __syncthreads();
  unsigned short* dst = out + ((size_t)b * T_ + t0) * C_ + c0;
#pragma unroll
  for (int i = 0; i < 32; i += 8)
    dst[(size_t)(y + i) * C_ + x] = f2bf(tile[x][y + i]);
}

// ---------------------------------------------------------------------------
// Projection GEMM: out[o][t] = sum_c W[o][c] * Xt[t][c] + bias[o]
// MODE 0: RoPE epilogue -> [b][h][t][d] bf16   (Q, K)
// MODE 1: transposed    -> [b][h][d][t] bf16   (V)
// MODE 2: plain f32     -> [b][o][t]           (O, = [B,C,T] output)
// 128 thr = 4 waves; block tile 64(M=o) x 64(N=t); wave 64x16; K-step 32.
// Double-buffered LDS; B tile staged with async LDS DMA.
// ---------------------------------------------------------------------------
template <int MODE>
__global__ __launch_bounds__(128) void proj_kernel(
    const float* __restrict__ W, const unsigned short* __restrict__ Xt,
    const float* __restrict__ bias, void* __restrict__ outp) {
  __shared__ alignas(16) unsigned short ldsA[2][64][40];  // W tile  [o][c] bf16
  __shared__ alignas(16) unsigned short ldsB[2][64][40];  // Xt tile [t][c] bf16
  const int b = blockIdx.z;
  const int nBase = blockIdx.x * 64;                      // t
  const int mBase = blockIdx.y * 64;                      // o (one head per tile)
  const int tid = threadIdx.x;
  const int w = tid >> 5, lane = tid & 31, half = lane >> 4, lr = lane & 15;
  const int koff = half * 8;

  FAcc acc[4];
#pragma unroll
  for (int ms = 0; ms < 4; ++ms)
#pragma unroll
    for (int r = 0; r < 8; ++r) acc[ms].f[r] = 0.0f;

  auto stage = [&](int k, int buf) {
    // A: W[mBase..+64][k..+32] f32 -> bf16 (needs convert: VGPR-staged)
#pragma unroll
    for (int q = 0; q < 4; ++q) {
      const int idx = tid * 4 + q;                        // 512 float4
      const int row = idx >> 3, col = (idx & 7) * 4;
      const float4 f = *(const float4*)(W + (size_t)(mBase + row) * C_ + k + col);
      *(unsigned int*)&ldsA[buf][row][col]     = f2bf2(f.x, f.y);
      *(unsigned int*)&ldsA[buf][row][col + 2] = f2bf2(f.z, f.w);
    }
    // B: Xt[b][nBase..+64][k..+32] bf16: async memory->LDS
#pragma unroll
    for (int q = 0; q < 2; ++q) {
      const int idx = tid * 2 + q;                        // 256 x 16B
      const int row = idx >> 2, col = (idx & 3) * 8;
      async_b128(Xt + ((size_t)b * T_ + nBase + row) * C_ + k + col,
                 &ldsB[buf][row][col]);
    }
  };

  stage(0, 0);
  for (int k = 0; k < C_; k += 32) {
    const int buf = (k >> 5) & 1;
    wait_async();
    __syncthreads();
    if (k + 32 < C_) stage(k + 32, buf ^ 1);

    BFrag bf;                                             // shared by 4 M-subtiles
    bf.q[0] = *(const uint4*)&ldsB[buf][w * 16 + lr][half * 8];
    bf.q[1] = *(const uint4*)&ldsB[buf][w * 16 + lr][16 + half * 8];
#pragma unroll
    for (int ms = 0; ms < 4; ++ms) {
      BFrag af;
      af.q[0] = *(const uint4*)&ldsA[buf][ms * 16 + lr][koff];
      af.q[1] = *(const uint4*)&ldsA[buf][ms * 16 + lr][16 + koff];
      acc[ms].v = __builtin_amdgcn_wmma_f32_16x16x32_bf16(
          false, af.v, false, bf.v, (short)0, acc[ms].v, false, false);
    }
  }

  // ---- epilogue ----
  const int t = nBase + w * 16 + lr;
  float val[4][8];
#pragma unroll
  for (int ms = 0; ms < 4; ++ms)
#pragma unroll
    for (int r = 0; r < 8; ++r)
      val[ms][r] = acc[ms].f[r] + bias[mBase + ms * 16 + half * 8 + r];

  if (MODE == 2) {
    float* out = (float*)outp;
#pragma unroll
    for (int ms = 0; ms < 4; ++ms)
#pragma unroll
      for (int r = 0; r < 8; ++r) {
        const int o = mBase + ms * 16 + half * 8 + r;
        out[((size_t)b * C_ + o) * T_ + t] = val[ms][r];
      }
  } else if (MODE == 0) {
    // RoPE: pair (d, d+16), d in [0,16) -> acc pair (ms=0, ms=1), same lane/reg
    const int h = mBase >> 6;
#pragma unroll
    for (int r = 0; r < 8; ++r) {
      const int d0 = half * 8 + r;
      const float th = __expf(-(float)d0 * (9.2103403719761836f / 16.0f)); // 10000^(-d0/16)
      const float ang = (float)t * th;
      float sn, cs; __sincosf(ang, &sn, &cs);
      const float v0 = val[0][r], v1 = val[1][r];
      val[0][r] = v0 * cs - v1 * sn;
      val[1][r] = v1 * cs + v0 * sn;
    }
    unsigned short* out = (unsigned short*)outp;
    const size_t base = (((size_t)b * H_ + h) * T_ + t) * D_;
#pragma unroll
    for (int ms = 0; ms < 4; ++ms) {
      union { unsigned int u[4]; uint4 q; } pk;
#pragma unroll
      for (int r = 0; r < 4; ++r)
        pk.u[r] = f2bf2(val[ms][2 * r], val[ms][2 * r + 1]);
      *(uint4*)(out + base + ms * 16 + half * 8) = pk.q;  // 16B store
    }
  } else {  // MODE 1: V transposed -> [b][h][d][t]
    const int h = mBase >> 6;
    unsigned short* out = (unsigned short*)outp;
#pragma unroll
    for (int ms = 0; ms < 4; ++ms)
#pragma unroll
      for (int r = 0; r < 8; ++r) {
        const int d = ms * 16 + half * 8 + r;
        out[(((size_t)b * H_ + h) * D_ + d) * T_ + t] = f2bf(val[ms][r]);
      }
  }
}

// ---------------------------------------------------------------------------
// Flash attention, one (b,h): block = 64 q rows, 4 waves x 16 rows.
// scores = QK^T/8 - log1p(|tk-tq|), mask, online softmax, O += P*V.
// K/V^T tiles double-buffered via async LDS DMA.
// ---------------------------------------------------------------------------
__global__ __launch_bounds__(128) void attn_kernel(
    const unsigned short* __restrict__ Qh, const unsigned short* __restrict__ Kh,
    const unsigned short* __restrict__ Vt, const int* __restrict__ mask,
    unsigned short* __restrict__ OT) {
  __shared__ alignas(16) unsigned short ldsK[2][64][72];   // K rows [t][d]
  __shared__ alignas(16) unsigned short ldsV[2][64][72];   // V^T   [d][t]
  __shared__ alignas(16) unsigned short ldsP[4][16][72];   // per-wave probs
  const int b = blockIdx.z, h = blockIdx.y;
  const int tid = threadIdx.x;
  const int w = tid >> 5, lane = tid & 31, half = lane >> 4, lr = lane & 15;
  const int koff = half * 8;
  const int qb = blockIdx.x * 64 + w * 16;

  const size_t headQK = ((size_t)b * H_ + h) * T_ * D_;
  const size_t headV  = ((size_t)b * H_ + h) * (size_t)D_ * T_;

  BFrag qf[2];                                             // Q row, all 64 d
  {
    const unsigned short* qrow = Qh + headQK + (size_t)(qb + lr) * D_;
    qf[0].q[0] = *(const uint4*)(qrow + koff);
    qf[0].q[1] = *(const uint4*)(qrow + 16 + koff);
    qf[1].q[0] = *(const uint4*)(qrow + 32 + koff);
    qf[1].q[1] = *(const uint4*)(qrow + 48 + koff);
  }

  float tqf[8];
#pragma unroll
  for (int r = 0; r < 8; ++r) tqf[r] = (float)(qb + half * 8 + r);

  float mrow[8], lrow[8];
  FAcc oacc[4];
#pragma unroll
  for (int r = 0; r < 8; ++r) { mrow[r] = -1e30f; lrow[r] = 0.0f; }
#pragma unroll
  for (int dt = 0; dt < 4; ++dt)
#pragma unroll
    for (int r = 0; r < 8; ++r) oacc[dt].f[r] = 0.0f;

  auto stage = [&](int kb, int buf) {
#pragma unroll
    for (int q = 0; q < 4; ++q) {
      const int idx = tid * 4 + q;                         // 512 x 16B per tile
      const int row = idx >> 3, col = (idx & 7) * 8;
      async_b128(Kh + headQK + (size_t)(kb + row) * D_ + col, &ldsK[buf][row][col]);
      async_b128(Vt + headV + (size_t)row * T_ + kb + col,   &ldsV[buf][row][col]);
    }
  };

  stage(0, 0);
  for (int kb = 0; kb < T_; kb += 64) {
    const int buf = (kb >> 6) & 1;
    wait_async();
    __syncthreads();
    if (kb + 64 < T_) stage(kb + 64, buf ^ 1);

    FAcc s[4];
#pragma unroll
    for (int nt = 0; nt < 4; ++nt) {                       // S = Q K^T
#pragma unroll
      for (int r = 0; r < 8; ++r) s[nt].f[r] = 0.0f;
#pragma unroll
      for (int st = 0; st < 2; ++st) {
        BFrag bf;
        bf.q[0] = *(const uint4*)&ldsK[buf][nt * 16 + lr][st * 32 + half * 8];
        bf.q[1] = *(const uint4*)&ldsK[buf][nt * 16 + lr][st * 32 + 16 + half * 8];
        s[nt].v = __builtin_amdgcn_wmma_f32_16x16x32_bf16(
            false, qf[st].v, false, bf.v, (short)0, s[nt].v, false, false);
      }
    }

    float rowmax[8];
#pragma unroll
    for (int r = 0; r < 8; ++r) rowmax[r] = -1e30f;
#pragma unroll
    for (int nt = 0; nt < 4; ++nt) {
      const int tk = kb + nt * 16 + lr;
      const int mv = mask[(size_t)b * T_ + tk];
      const float tkf = (float)tk;
#pragma unroll
      for (int r = 0; r < 8; ++r) {
        float sv = s[nt].f[r] * 0.125f - log1p_abs(tkf - tqf[r]);
        sv = (mv == 0) ? -10000.0f : sv;
        s[nt].f[r] = sv;
        rowmax[r] = fmaxf(rowmax[r], sv);
      }
    }
#pragma unroll
    for (int r = 0; r < 8; ++r) {                          // row reduce in 16-lane half
      float rm = rowmax[r];
      rm = fmaxf(rm, __shfl_xor(rm, 1, 32));
      rm = fmaxf(rm, __shfl_xor(rm, 2, 32));
      rm = fmaxf(rm, __shfl_xor(rm, 4, 32));
      rm = fmaxf(rm, __shfl_xor(rm, 8, 32));
      rowmax[r] = rm;
    }
    float corr[8], rsum[8];
#pragma unroll
    for (int r = 0; r < 8; ++r) {
      const float mnew = fmaxf(mrow[r], rowmax[r]);
      corr[r] = __expf(mrow[r] - mnew);
      mrow[r] = mnew;
      rsum[r] = 0.0f;
    }
#pragma unroll
    for (int nt = 0; nt < 4; ++nt)                         // P = exp(S-m) -> LDS
#pragma unroll
      for (int r = 0; r < 8; ++r) {
        const float pv = __expf(s[nt].f[r] - mrow[r]);
        rsum[r] += pv;
        ldsP[w][half * 8 + r][nt * 16 + lr] = f2bf(pv);
      }
#pragma unroll
    for (int r = 0; r < 8; ++r) {
      float rs = rsum[r];
      rs += __shfl_xor(rs, 1, 32);
      rs += __shfl_xor(rs, 2, 32);
      rs += __shfl_xor(rs, 4, 32);
      rs += __shfl_xor(rs, 8, 32);
      lrow[r] = lrow[r] * corr[r] + rs;
#pragma unroll
      for (int dt = 0; dt < 4; ++dt) oacc[dt].f[r] *= corr[r];
    }

    asm volatile("s_wait_dscnt 0x0" ::: "memory");         // same-wave P RAW

#pragma unroll
    for (int dt = 0; dt < 4; ++dt)                         // O += P V
#pragma unroll
      for (int st = 0; st < 2; ++st) {
        BFrag af, bf;
        af.q[0] = *(const uint4*)&ldsP[w][lr][st * 32 + koff];
        af.q[1] = *(const uint4*)&ldsP[w][lr][st * 32 + 16 + koff];
        bf.q[0] = *(const uint4*)&ldsV[buf][dt * 16 + lr][st * 32 + half * 8];
        bf.q[1] = *(const uint4*)&ldsV[buf][dt * 16 + lr][st * 32 + 16 + half * 8];
        oacc[dt].v = __builtin_amdgcn_wmma_f32_16x16x32_bf16(
            false, af.v, false, bf.v, (short)0, oacc[dt].v, false, false);
      }
  }

#pragma unroll
  for (int r = 0; r < 8; ++r) {
#if __has_builtin(__builtin_amdgcn_rcpf)
    const float inv = __builtin_amdgcn_rcpf(lrow[r]);
#else
    const float inv = 1.0f / lrow[r];
#endif
    const int t = qb + half * 8 + r;
#pragma unroll
    for (int dt = 0; dt < 4; ++dt) {
      const int c = h * 64 + dt * 16 + lr;
      OT[((size_t)b * T_ + t) * C_ + c] = f2bf(oacc[dt].f[r] * inv);
    }
  }
}

// ---------------------------------------------------------------------------
extern "C" void kernel_launch(void* const* d_in, const int* in_sizes, int n_in,
                              void* d_out, int out_size, void* d_ws, size_t ws_size,
                              hipStream_t stream) {
  (void)in_sizes; (void)n_in; (void)out_size; (void)ws_size;
  const float* x   = (const float*)d_in[0];
  const float* ctx = (const float*)d_in[1];
  const int* amask = (const int*)d_in[2];
  const float* Wq = (const float*)d_in[3]; const float* bq = (const float*)d_in[4];
  const float* Wk = (const float*)d_in[5]; const float* bk = (const float*)d_in[6];
  const float* Wv = (const float*)d_in[7]; const float* bv = (const float*)d_in[8];
  const float* Wo = (const float*)d_in[9]; const float* bo = (const float*)d_in[10];

  const size_t EL = (size_t)B_ * T_ * C_;                 // 8M elements
  unsigned short* xT = (unsigned short*)d_ws;             // bf16 scratch
  unsigned short* cT = xT + EL;
  unsigned short* Qh = cT + EL;
  unsigned short* Kh = Qh + EL;
  unsigned short* Vt = Kh + EL;
  unsigned short* OT = xT;                                // alias: xT dead after Q

  dim3 tb(32, 8), tg(T_ / 32, C_ / 32, B_);
  transpose_bf16_kernel<<<tg, tb, 0, stream>>>(x, xT);
  transpose_bf16_kernel<<<tg, tb, 0, stream>>>(ctx, cT);

  dim3 pg(T_ / 64, C_ / 64, B_);
  proj_kernel<0><<<pg, 128, 0, stream>>>(Wq, xT, bq, Qh);
  proj_kernel<0><<<pg, 128, 0, stream>>>(Wk, cT, bk, Kh);
  proj_kernel<1><<<pg, 128, 0, stream>>>(Wv, cT, bv, Vt);

  dim3 ag(T_ / 64, H_, B_);
  attn_kernel<<<ag, 128, 0, stream>>>(Qh, Kh, Vt, amask, OT);

  proj_kernel<2><<<pg, 128, 0, stream>>>(Wo, OT, bo, d_out);
}
